// ContrastiveCriterion_79594333929503
// MI455X (gfx1250) — matrix operand order repaired and checked by reading, not compiled, not run on hardware
//
#include <hip/hip_runtime.h>
#include <math.h>

typedef __attribute__((ext_vector_type(16))) _Float16 v16h;
typedef __attribute__((ext_vector_type(8)))  _Float16 v8h;
typedef __attribute__((ext_vector_type(8)))  float    v8f;

#define BB 2
#define NN 4096
#define DD 512
#define CS 4            // column chunks per row strip (occupancy)
#define NCHUNK (NN / CS)

// ---------------------------------------------------------------------------
// Kernel 1: normalize text rows, gather+normalize point rows, exact f32 diag,
// zero rowsum / colsum / ranksum accumulators.
// grid = B*N blocks of 128 threads (each thread owns 4 of the 512 dims)
// ---------------------------------------------------------------------------
__global__ void __launch_bounds__(128) prep_kernel(
    const float* __restrict__ ep, const float* __restrict__ et,
    const int* __restrict__ idx,
    _Float16* __restrict__ keyH, _Float16* __restrict__ refH,
    float* __restrict__ diag, float* __restrict__ rowsum,
    float* __restrict__ colsum, float* __restrict__ ranksum)
{
  const int row = blockIdx.x;            // 0 .. B*N-1
  const int b   = row / NN;
  const int t   = threadIdx.x;           // 128 threads
  const int src = idx[row];              // gather index within sample b

  const float* kp = et + (size_t)row * DD;
  const float* rp = ep + ((size_t)b * NN + (size_t)src) * DD;

  float kv[4], rv[4];
  float ssk = 0.f, ssr = 0.f, dt = 0.f;
#pragma unroll
  for (int u = 0; u < 4; ++u) {
    const int d = t + u * 128;
    kv[u] = kp[d];
    rv[u] = rp[d];
    ssk += kv[u] * kv[u];
    ssr += rv[u] * rv[u];
    dt  += kv[u] * rv[u];
  }
  // wave32 reduction
#pragma unroll
  for (int off = 16; off; off >>= 1) {
    ssk += __shfl_xor(ssk, off, 32);
    ssr += __shfl_xor(ssr, off, 32);
    dt  += __shfl_xor(dt,  off, 32);
  }
  __shared__ float sK[4], sR[4], sD[4];
  const int w = t >> 5, ln = t & 31;
  if (ln == 0) { sK[w] = ssk; sR[w] = ssr; sD[w] = dt; }
  __syncthreads();
  ssk = sK[0] + sK[1] + sK[2] + sK[3];
  ssr = sR[0] + sR[1] + sR[2] + sR[3];
  dt  = sD[0] + sD[1] + sD[2] + sD[3];

  const float invk = 1.0f / fmaxf(sqrtf(ssk), 1e-12f);
  const float invr = 1.0f / fmaxf(sqrtf(ssr), 1e-12f);

#pragma unroll
  for (int u = 0; u < 4; ++u) {
    const int d = t + u * 128;
    keyH[(size_t)row * DD + d] = (_Float16)(kv[u] * invk);
    refH[(size_t)row * DD + d] = (_Float16)(rv[u] * invr);
  }
  if (t == 0) {
    diag[row]   = dt * invk * invr;   // exact f32 diagonal similarity
    rowsum[row] = 0.0f;
    colsum[row] = 0.0f;
    if (row == 0) ranksum[0] = 0.0f;
  }
}

// ---------------------------------------------------------------------------
// Kernel 2: fused WMMA sim + softmax-sumexp (rows & cols) + relu rank sum.
// 1024 waves: each owns 32 rows (two 16-row A tiles resident in VGPRs) and a
// 1024-column chunk. Each B fragment feeds TWO WMMAs (register reuse), and a
// 4-deep rotating register buffer keeps 8 b128 loads in flight; the last 4
// prefetches of a tile already target the NEXT column tile so the epilogue
// (exp/shuffle/atomic) overlaps the next tile's load latency.
// ---------------------------------------------------------------------------
__global__ void __launch_bounds__(256) sim_kernel(
    const _Float16* __restrict__ keyH, const _Float16* __restrict__ refH,
    const float* __restrict__ diag, const float* __restrict__ logit_scale,
    float* __restrict__ rowsum, float* __restrict__ colsum,
    float* __restrict__ ranksum)
{
  const int lane = threadIdx.x & 31;
  const int wave = (int)((blockIdx.x * blockDim.x + threadIdx.x) >> 5);
  const int b    = wave >> 9;                  // /(128 strips * 4 chunks)
  const int rem  = wave & 511;
  const int i0   = (rem >> 2) * 32;            // first of 32 rows
  const int n_st = (rem & 3) * NCHUNK;         // column chunk start
  const int lm   = lane & 15;
  const int h    = lane >> 4;
  const float s  = logit_scale[0];

  // --- resident A fragments: rows i0..i0+31, K = 0..511 -------------------
  // A layout (16x32), matching B: lanes 0-15 hold K = kb+[0..15];
  // lanes 16-31 hold K = kb+16+[0..15].
  v16h afrag[2][16];
#pragma unroll
  for (int tt = 0; tt < 2; ++tt) {
    const _Float16* arow =
        refH + ((size_t)b * NN + (size_t)(i0 + 16 * tt + lm)) * DD;
#pragma unroll
    for (int kk = 0; kk < 16; ++kk) {
      const int kb = kk * 32;
      const v8h lo = *(const v8h*)(arow + kb + 16 * h);
      const v8h hi = *(const v8h*)(arow + kb + 16 * h + 8);
      v16h a;
#pragma unroll
      for (int e = 0; e < 8; ++e) { a[e] = lo[e]; a[8 + e] = hi[e]; }
      afrag[tt][kk] = a;
    }
  }

  float rowacc[2][8];
#pragma unroll
  for (int tt = 0; tt < 2; ++tt)
#pragma unroll
    for (int r = 0; r < 8; ++r) rowacc[tt][r] = 0.f;
  float rk = 0.f;

  // B layout (32x16): lane holds key[n][kb + 16*h .. +15] (contiguous 32B)
  const _Float16* kb0   = keyH + (size_t)b * NN * DD + (size_t)lm * DD + 16 * h;
  const float*    dbase = diag + (size_t)b * NN;

  // prime 4-deep B prefetch buffer with tile n_st, kk = 0..3
  v16h bbuf[4];
  {
    const _Float16* brow = kb0 + (size_t)n_st * DD;
#pragma unroll
    for (int p = 0; p < 4; ++p) bbuf[p] = *(const v16h*)(brow + p * 32);
  }

  for (int n0 = n_st; n0 < n_st + NCHUNK; n0 += 16) {
    const _Float16* brow = kb0 + (size_t)n0 * DD;
    const bool last      = (n0 + 16 >= n_st + NCHUNK);
    // next tile base (re-read current tile harmlessly on the last iteration)
    const _Float16* brown = last ? brow : (brow + (size_t)16 * DD);

    v8f acc0 = {};
    v8f acc1 = {};
#pragma unroll
    for (int kk = 0; kk < 16; ++kk) {
      const v16h bcur = bbuf[kk & 3];
      if (kk < 12) {
        bbuf[kk & 3] = *(const v16h*)(brow + (kk + 4) * 32);   // same tile
      } else {
        bbuf[kk & 3] = *(const v16h*)(brown + (kk - 12) * 32); // next tile
      }
      acc0 = __builtin_amdgcn_wmma_f32_16x16x32_f16(
          false, afrag[0][kk], false, bcur, (short)0, acc0, false, false);
      acc1 = __builtin_amdgcn_wmma_f32_16x16x32_f16(
          false, afrag[1][kk], false, bcur, (short)0, acc1, false, false);
    }

    // epilogue: accT[r] is sim(m = i0 + 16*T + r + 8*h, n = n0 + lm)
    const float dcol = dbase[n0 + lm];
    float colpart = 0.f;
#pragma unroll
    for (int r = 0; r < 8; ++r) {
      const float sv0 = acc0[r];
      const float sv1 = acc1[r];
      const float e0  = __expf(s * sv0);  // |s*sv| <= ~2.7: no max-shift needed
      const float e1  = __expf(s * sv1);
      rowacc[0][r] += e0;
      rowacc[1][r] += e1;
      colpart      += e0 + e1;
      rk += fmaxf(sv0 - dcol, 0.f);       // relu(sim[i,j] - sim[j,j])
      rk += fmaxf(sv1 - dcol, 0.f);
    }
    colpart += __shfl_xor(colpart, 16, 32);          // combine both row halves
    if (h == 0) atomicAdd(&colsum[(size_t)b * NN + n0 + lm], colpart);
  }

  // rowsum partials (this wave only covered NCHUNK columns): atomic combine
#pragma unroll
  for (int tt = 0; tt < 2; ++tt) {
#pragma unroll
    for (int r = 0; r < 8; ++r) {
      float v = rowacc[tt][r];
      v += __shfl_xor(v, 8, 32);
      v += __shfl_xor(v, 4, 32);
      v += __shfl_xor(v, 2, 32);
      v += __shfl_xor(v, 1, 32);
      if (lm == 0)
        atomicAdd(&rowsum[(size_t)b * NN + i0 + 16 * tt + r + 8 * h], v);
    }
  }

  // rank partial: full-wave reduce, one atomic per wave
  rk += __shfl_xor(rk, 16, 32);
  rk += __shfl_xor(rk, 8, 32);
  rk += __shfl_xor(rk, 4, 32);
  rk += __shfl_xor(rk, 2, 32);
  rk += __shfl_xor(rk, 1, 32);
  if (lane == 0) atomicAdd(ranksum, rk);
}

// ---------------------------------------------------------------------------
// Kernel 3: final weighted CE mean + rank scaling. One block.
// ---------------------------------------------------------------------------
__global__ void __launch_bounds__(256) finish_kernel(
    const float* __restrict__ rowsum, const float* __restrict__ colsum,
    const float* __restrict__ diag, const float* __restrict__ dn,
    const float* __restrict__ logit_scale, const float* __restrict__ ranksum,
    float* __restrict__ out)
{
  const float s = logit_scale[0];
  float acc = 0.f;
  for (int i = threadIdx.x; i < BB * NN; i += blockDim.x) {
    const float d   = diag[i];
    const float cep = logf(rowsum[i]) - s * d;   // -log_softmax row diag
    const float cet = logf(colsum[i]) - s * d;   // -log_softmax col diag
    acc += 0.5f * (cep + cet) * dn[i];
  }
#pragma unroll
  for (int off = 16; off; off >>= 1) acc += __shfl_xor(acc, off, 32);
  __shared__ float sm[8];
  const int w = threadIdx.x >> 5, ln = threadIdx.x & 31;
  if (ln == 0) sm[w] = acc;
  __syncthreads();
  if (threadIdx.x == 0) {
    float t = 0.f;
#pragma unroll
    for (int k = 0; k < 8; ++k) t += sm[k];
    out[0] = 1.0f * t / (float)(BB * NN);        // CONTRASTIVE_W * mean
    out[1] = 0.5f * ranksum[0] / (float)BB;      // RANKING_W * mean
  }
}

// ---------------------------------------------------------------------------
extern "C" void kernel_launch(void* const* d_in, const int* in_sizes, int n_in,
                              void* d_out, int out_size, void* d_ws, size_t ws_size,
                              hipStream_t stream) {
  (void)in_sizes; (void)n_in; (void)out_size; (void)ws_size;
  const float* ep  = (const float*)d_in[0];   // emb_point  [B,N,D] f32
  const float* et  = (const float*)d_in[1];   // emb_text   [B,N,D] f32
  const float* dn  = (const float*)d_in[2];   // dist_norm  [B,N]   f32
  const int*   idx = (const int*)  d_in[3];   // pos_idx    [B,N]   i32
  const float* ls  = (const float*)d_in[4];   // logit_scale scalar f32
  float* out = (float*)d_out;

  char* ws = (char*)d_ws;
  size_t off = 0;
  auto carve = [&](size_t bytes) -> void* {
    void* p = ws + off;
    off += (bytes + 255) & ~(size_t)255;
    return p;
  };
  _Float16* keyH    = (_Float16*)carve((size_t)BB * NN * DD * sizeof(_Float16));
  _Float16* refH    = (_Float16*)carve((size_t)BB * NN * DD * sizeof(_Float16));
  float*    diag    = (float*)   carve((size_t)BB * NN * sizeof(float));
  float*    rowsum  = (float*)   carve((size_t)BB * NN * sizeof(float));
  float*    colsum  = (float*)   carve((size_t)BB * NN * sizeof(float));
  float*    ranksum = (float*)   carve(256);

  prep_kernel<<<dim3(BB * NN), dim3(128), 0, stream>>>(
      ep, et, idx, keyH, refH, diag, rowsum, colsum, ranksum);

  // 1024 waves (2 samples * 128 row-strips * 4 column chunks), 8 waves/block
  sim_kernel<<<dim3((BB * (NN / 32) * CS * 32) / 256), dim3(256), 0, stream>>>(
      keyH, refH, diag, ls, rowsum, colsum, ranksum);

  finish_kernel<<<dim3(1), dim3(256), 0, stream>>>(
      rowsum, colsum, diag, dn, ls, ranksum, out);
}
